// SemiCRF_81896436400532
// MI455X (gfx1250) — matrix operand order repaired and checked by reading, not compiled, not run on hardware
//
#include <hip/hip_runtime.h>

typedef __attribute__((ext_vector_type(16))) _Float16     v16h;
typedef __attribute__((ext_vector_type(8)))  float        v8f;
typedef __attribute__((ext_vector_type(4)))  unsigned int u32x4;
typedef __attribute__((ext_vector_type(8)))  int          i32x8;
typedef __attribute__((ext_vector_type(4)))  int          i32x4;

#define NEGINF (-1e30f)

constexpr int B = 8, S = 512, L = 32, M = 64;
constexpr int TDM_MIN_FEND = 63;   // first f_end whose whole 64-row window has start >= 0

// Single shared struct so the TDM lds_addr is a known static offset.
// segbuf MUST be the first member (LDS offset 0 / 8192 for the two parities).
struct Smem {
    float    segbuf[2][M][L];  // double-buffered seg window; reused in-place for scores
    float    alphaS[M][L];     // ring buffer: slot (p & 63) holds log-alpha at frame p
    _Float16 Xh[M][L];         // exp(alpha - rowmax), f16 WMMA A-matrix
    _Float16 Eh[L][L];         // exp(transitions), f16 WMMA B-matrix (loop invariant)
    float    c1[M];            // per-row max of alpha window
    float    diagS[L];
    float    initS[L];
    float    mxS[L];
    float    part[8][L];
};

// Issue a TDM 2D tile load: 64 rows x 32 floats, row stride S*L elements,
// global -> LDS at byte offset lds_off. Descriptor per CDNA5 ISA ch. 8.
__device__ __forceinline__ void issue_tdm_load(const float* gptr, unsigned lds_off) {
    unsigned long long ga = (unsigned long long)(uintptr_t)gptr;
    u32x4 g0;
    g0[0] = 1u;                                               // count=1, user descriptor
    g0[1] = lds_off;                                          // lds_addr (bytes)
    g0[2] = (unsigned)(ga & 0xFFFFFFFFu);                     // global_addr[31:0]
    g0[3] = (unsigned)((ga >> 32) & 0x1FFFFFFu) | (2u << 30); // global_addr[56:32] | type=2
    i32x8 g1;
    g1[0] = (2 << 16);          // workgroup_mask=0 | data_size=2 (4 bytes)
    g1[1] = (int)(L << 16);     // atomic_barrier_addr=0 | tensor_dim0[15:0]=32
    g1[2] = (int)(M << 16);     // tensor_dim0[31:16]=0  | tensor_dim1[15:0]=64
    g1[3] = (int)(L << 16);     // tensor_dim1[31:16]=0  | tile_dim0=32
    g1[4] = M;                  // tile_dim1=64 | tile_dim2=0
    g1[5] = S * L;              // tensor_dim0_stride = 16384 elements
    g1[6] = 0;                  // stride hi | tensor_dim1_stride lo (unused, 2D tile)
    g1[7] = 0;
    i32x4 gz = {0, 0, 0, 0};
#if defined(__clang_major__) && (__clang_major__ >= 23)
    i32x8 gz8 = {0, 0, 0, 0, 0, 0, 0, 0};
    __builtin_amdgcn_tensor_load_to_lds(g0, g1, gz, gz, gz8, 0);
#else
    __builtin_amdgcn_tensor_load_to_lds(g0, g1, gz, gz, 0);
#endif
}

__global__ __launch_bounds__(256)
void semicrf_fwd_kernel(const float* __restrict__ seg,
                        const float* __restrict__ trans,
                        float* __restrict__ out)
{
    __shared__ Smem sm;

    const int b     = blockIdx.x;
    const int tid   = threadIdx.x;
    const int lane  = tid & 31;
    const int wv    = tid >> 5;     // wave 0..7
    const int rt    = wv >> 1;      // row tile (w-chunk) 0..3
    const int ct    = wv & 1;       // col tile (cur-chunk) 0..1
    const int mhalf = lane >> 4;    // lane half-select
    const int l15   = lane & 15;

    // ---- one-time init: exp(transitions), diagonal, NEG-filled alpha ring ----
    for (int i = tid; i < L * L; i += 256) {
        int p = i >> 5, c = i & 31;
        sm.Eh[p][c] = (_Float16)__expf(trans[p * L + c]);
    }
    for (int i = tid; i < M * L; i += 256)
        sm.alphaS[i >> 5][i & 31] = NEGINF;
    if (tid < L) sm.diagS[tid] = trans[tid * L + tid];
    __syncthreads();

    // alpha at frame 0 = segment_score[b, 0, 0, :]
    if (tid < L)
        sm.alphaS[0][tid] = seg[(size_t)b * S * S * L + tid];

    // Loop-invariant B fragment: E[k=prev][n=cur] (32x16 f16 tile).
    // B layout: lanes 0-15 hold K=0..7,16..23, lanes 16-31 K=8..15,24..31 (matches A).
    v16h bfrag;
    {
        int n = ct * 16 + l15;
        #pragma unroll
        for (int v = 0; v < 8; ++v) {
            int k = ((v & 4) ? 16 : 0) + mhalf * 8 + (v & 3) * 2;
            bfrag[2 * v]     = sm.Eh[k][n];
            bfrag[2 * v + 1] = sm.Eh[k + 1][n];
        }
    }

    // Pre-fill the seg window for f_end = 1 (guarded VMEM path).
    {
        const int fe = 1, buf = fe & 1;
        for (int i = tid; i < M * L; i += 256) {
            int w = i >> 5, c = i & 31;
            int start = fe - M + 1 + w;
            float v = 0.f;
            if (start >= 0) v = seg[(((size_t)b * S + start) * S + fe) * L + c];
            sm.segbuf[buf][w][c] = v;
        }
    }
    __syncthreads();

    // ============== sequential scan over end frames ==============
    for (int f_end = 1; f_end < S; ++f_end) {
        const int p = f_end & 1;

        // ---- issue next step's window load into the other buffer ----
        if (f_end + 1 < S) {
            const int fe2 = f_end + 1, nb = p ^ 1;
            if (fe2 >= TDM_MIN_FEND) {
                // whole window in-bounds: async TDM tile load, overlapped with compute
                if (wv == 0) {
                    const int start0 = fe2 - M + 1;
                    issue_tdm_load(&seg[(((size_t)b * S + start0) * S + fe2) * L],
                                   (unsigned)(nb * (M * L * (int)sizeof(float))));
                }
            } else {
                // early steps: guarded VMEM fill (rows with start<0 zeroed; the
                // -1e30 alpha guard kills those paths anyway)
                for (int i = tid; i < M * L; i += 256) {
                    int w = i >> 5, c = i & 31;
                    int start = fe2 - M + 1 + w;
                    float v = 0.f;
                    if (start >= 0) v = seg[(((size_t)b * S + start) * S + fe2) * L + c];
                    sm.segbuf[nb][w][c] = v;
                }
            }
        }

        // ---- per-row alpha maxes ----
        if (tid < M) {
            int slot = (f_end + tid) & (M - 1);   // frame f_end - M + tid (mod ring)
            float m = sm.alphaS[slot][0];
            #pragma unroll 4
            for (int q = 1; q < L; ++q) m = fmaxf(m, sm.alphaS[slot][q]);
            sm.c1[tid] = m;
        }
        __syncthreads();

        // ---- Xh = exp(alpha - rowmax) in f16; init-segment term ----
        for (int i = tid; i < M * L; i += 256) {
            int w = i >> 5, q = i & 31;
            int slot = (f_end + w) & (M - 1);
            sm.Xh[w][q] = (_Float16)__expf(sm.alphaS[slot][q] - sm.c1[w]);
        }
        if (tid < L) {
            float iv = NEGINF;
            if (f_end < M) {
                float fe = (float)f_end;
                // start = 0 row sits at window slot w = M-1-f_end
                iv = sm.segbuf[p][M - 1 - f_end][tid] * (fe + 1.f) + sm.diagS[tid] * fe;
            }
            sm.initS[tid] = iv;
        }
        __syncthreads();

        // ---- Y = X @ E via WMMA; log-domain continuation scores (in-place) ----
        v16h afrag;
        {
            int w = rt * 16 + l15;
            #pragma unroll
            for (int v = 0; v < 8; ++v) {
                int k0 = ((v & 4) ? 16 : 0) + mhalf * 8 + (v & 3) * 2;
                afrag[2 * v]     = sm.Xh[w][k0];
                afrag[2 * v + 1] = sm.Xh[w][k0 + 1];
            }
        }
        v8f acc = {};
        acc = __builtin_amdgcn_wmma_f32_16x16x32_f16(false, afrag, false, bfrag,
                                                     (short)0, acc, false, false);
        #pragma unroll
        for (int r = 0; r < 8; ++r) {
            int m   = r + mhalf * 8;          // D layout: VGPR r -> M = r + 8*half
            int w   = rt * 16 + m;
            int cur = ct * 16 + l15;
            float j  = (float)(M - w);
            float sc = __logf(acc[r]) + sm.c1[w]
                     + sm.segbuf[p][w][cur] * j + sm.diagS[cur] * (j - 1.f);
            sm.segbuf[p][w][cur] = sc;        // each (w,cur) owned by exactly one lane
        }
        __syncthreads();

        // ---- max over the 64 window slots (+ init) per label ----
        {
            int cur = tid & 31, chunk = tid >> 5;
            float m = NEGINF;
            #pragma unroll
            for (int k = 0; k < 8; ++k)
                m = fmaxf(m, sm.segbuf[p][chunk * 8 + k][cur]);
            sm.part[chunk][cur] = m;
        }
        __syncthreads();
        if (tid < L) {
            float m = sm.initS[tid];
            #pragma unroll
            for (int k = 0; k < 8; ++k) m = fmaxf(m, sm.part[k][tid]);
            sm.mxS[tid] = m;
        }
        __syncthreads();

        // ---- sum of exps; new alpha into the ring ----
        {
            int cur = tid & 31, chunk = tid >> 5;
            float mx = sm.mxS[cur];
            float s = 0.f;
            #pragma unroll
            for (int k = 0; k < 8; ++k)
                s += __expf(sm.segbuf[p][chunk * 8 + k][cur] - mx);
            sm.part[chunk][cur] = s;
        }
        __syncthreads();
        if (tid < L) {
            float mx = sm.mxS[tid];
            float s = __expf(sm.initS[tid] - mx);
            #pragma unroll
            for (int k = 0; k < 8; ++k) s += sm.part[k][tid];
            sm.alphaS[f_end & (M - 1)][tid] = mx + __logf(s);
        }

        // Drain the async tile before everyone crosses into the next step.
        // TENSORcnt is per-wave: NOP for waves that didn't issue.
        __builtin_amdgcn_s_wait_tensorcnt(0);
        __syncthreads();
    }

    // ---- final logsumexp over labels at frame S-1 ----
    if (tid == 0) {
        const float* a = sm.alphaS[(S - 1) & (M - 1)];
        float m = a[0];
        for (int q = 1; q < L; ++q) m = fmaxf(m, a[q]);
        float s = 0.f;
        for (int q = 0; q < L; ++q) s += __expf(a[q] - m);
        out[b] = m + __logf(s);
    }
}

extern "C" void kernel_launch(void* const* d_in, const int* in_sizes, int n_in,
                              void* d_out, int out_size, void* d_ws, size_t ws_size,
                              hipStream_t stream) {
    const float* segment_score = (const float*)d_in[0];   // [8, 512, 512, 32] f32
    const float* transitions   = (const float*)d_in[1];   // [32, 32] f32
    float* out = (float*)d_out;                           // [8] f32
    (void)in_sizes; (void)n_in; (void)out_size; (void)d_ws; (void)ws_size;
    semicrf_fwd_kernel<<<dim3(B), dim3(256), 0, stream>>>(segment_score, transitions, out);
}
